// GATNet_NC_51015621542151
// MI455X (gfx1250) — matrix operand order repaired and checked by reading, not compiled, not run on hardware
//
#include <hip/hip_runtime.h>

#define NNODES 30000
#define NEDGES 480000
#define ETOT   (NEDGES + NNODES)   // 510000, self loops appended
#define NHEAD  4
#define NCHAN  32
#define FDIM   128                 // H*C == D

typedef __attribute__((ext_vector_type(16))) _Float16 v16h;
typedef __attribute__((ext_vector_type(8)))  float    v8f;
typedef __attribute__((ext_vector_type(4)))  unsigned u32x4;
typedef __attribute__((ext_vector_type(8)))  int      i32x8;
typedef __attribute__((ext_vector_type(4)))  int      i32x4;

#ifndef __has_builtin
#define __has_builtin(x) 0
#endif

// ---------------------------------------------------------------------------
// Order-preserving float <-> unsigned key (for hardware atomic max on u32)
// ---------------------------------------------------------------------------
__device__ __forceinline__ unsigned fkey(float x) {
  unsigned u = __float_as_uint(x);
  return (u & 0x80000000u) ? ~u : (u | 0x80000000u);
}
__device__ __forceinline__ float funkey(unsigned k) {
  unsigned u = (k & 0x80000000u) ? (k & 0x7FFFFFFFu) : ~k;
  return __uint_as_float(u);
}

// ---------------------------------------------------------------------------
// f16 pre-conversion kernels (once per layer; halves GEMM global traffic)
// ---------------------------------------------------------------------------
__global__ __launch_bounds__(256) void cvt_x_f16(
    const float* __restrict__ X, _Float16* __restrict__ Xh)
{
  int i = blockIdx.x * blockDim.x + threadIdx.x;
  if (i < NNODES * FDIM) Xh[i] = (_Float16)X[i];
}

__global__ __launch_bounds__(256) void cvt_w_f16t(
    const float* __restrict__ W, _Float16* __restrict__ Wth)
{
  int i = blockIdx.x * blockDim.x + threadIdx.x;   // i = k*128 + n
  if (i < FDIM * FDIM) {
    int k = i >> 7, n = i & 127;
    Wth[n * FDIM + k] = (_Float16)W[i];            // store transposed [n][k]
  }
}

// ---------------------------------------------------------------------------
// TDM: issue a 2D tensor_load_to_lds (data_size = 2 bytes).
// D# group0: count=1 | lds_addr[63:32] | global_addr[120:64] | type=2
// D# group1: data_size@[17:16] | tensor_dim0@[79:48] | tensor_dim1@[111:80]
//            | tile_dim0@[127:112] | tile_dim1@[143:128] | stride0@[207:160]
// OOB rows (>= tensor_dim1) read back as zero -> free tail padding.
// This toolchain exposes the 6-arg builtin:
//   (uint32x4 g0, int32x8 g1, int32x4 g2, int32x4 g3, int32x8 extra, i32 cpol)
// ---------------------------------------------------------------------------
__device__ __forceinline__ void tdm_load_2d_f16(
    unsigned lds_addr, const void* gptr,
    unsigned td0, unsigned td1, unsigned stride0,
    unsigned tile0, unsigned tile1)
{
#if __has_builtin(__builtin_amdgcn_tensor_load_to_lds)
  unsigned long long ga = (unsigned long long)gptr;
  u32x4 g0;
  g0[0] = 1u;                                            // count=1 (valid D#)
  g0[1] = lds_addr;                                      // lds_addr (bytes)
  g0[2] = (unsigned)(ga & 0xFFFFFFFFull);                // global_addr[31:0]
  g0[3] = (unsigned)((ga >> 32) & 0x01FFFFFFull)         // global_addr[56:32]
          | (2u << 30);                                  // type=2 ("image")
  i32x8 g1;
  g1[0] = (int)(1u << 16);                               // data_size=1 -> 2B
  g1[1] = (int)((td0 & 0xFFFFu) << 16);                  // tensor_dim0 lo
  g1[2] = (int)((td0 >> 16) | ((td1 & 0xFFFFu) << 16));  // td0 hi | td1 lo
  g1[3] = (int)((td1 >> 16) | ((tile0 & 0xFFFFu) << 16));// td1 hi | tile_dim0
  g1[4] = (int)(tile1 & 0xFFFFu);                        // tile_dim1, tile_dim2=0
  g1[5] = (int)stride0;                                  // dim0 stride lo
  g1[6] = 0;                                             // stride hi | dim1 stride lo
  g1[7] = 0;
  i32x4 z4 = {0, 0, 0, 0};
  i32x8 z8 = {0, 0, 0, 0, 0, 0, 0, 0};
  __builtin_amdgcn_tensor_load_to_lds(g0, g1, z4, z4, z8, 0);
#else
  (void)lds_addr; (void)gptr; (void)td0; (void)td1;
  (void)stride0; (void)tile0; (void)tile1;
#endif
}

__device__ __forceinline__ void wait_tensorcnt0() {
#if __has_builtin(__builtin_amdgcn_s_wait_tensorcnt)
  __builtin_amdgcn_s_wait_tensorcnt(0);
#else
  asm volatile("s_wait_tensorcnt 0x0" ::: "memory");
#endif
}

// ---------------------------------------------------------------------------
// WMMA GEMM: Hout[nrows,128] = Xh[nrows,128] @ W[128,128]  (f16 in, f32 out)
// block = 128 threads (4 waves); tile = 64 rows x 128 cols.
// Tiles staged to LDS by the Tensor Data Mover (wave 0 issues, all consume).
// Wth already transposed [n][k] so each lane's B fragment is two contiguous 8-f16
// runs (lanes<16: K 0..7 & 16..23, lanes>=16: K 8..15 & 24..31, as for A).
// ---------------------------------------------------------------------------
__global__ __launch_bounds__(128) void gemm_wmma(
    const _Float16* __restrict__ Xh, const _Float16* __restrict__ Wth,
    float* __restrict__ Hout, int nrows)
{
  __shared__ _Float16 Xs[64 * 128];    // [row][k]
  __shared__ _Float16 Ws[128 * 128];   // [n][k]  (already transposed)

  const int t = threadIdx.x;
  const int rowBase = blockIdx.x * 64;

#if __has_builtin(__builtin_amdgcn_tensor_load_to_lds)
  if (t < 32) {                        // wave 0 drives both DMAs
    unsigned ldsX = (unsigned)(unsigned long long)(const void*)Xs;
    unsigned ldsW = (unsigned)(unsigned long long)(const void*)Ws;
    // X tile: 64x128 of a [nrows,128] tensor; ragged tail zero-filled by TDM
    tdm_load_2d_f16(ldsX, Xh + (size_t)rowBase * FDIM,
                    FDIM, (unsigned)(nrows - rowBase), FDIM, FDIM, 64);
    // W tile: full 128x128
    tdm_load_2d_f16(ldsW, Wth, FDIM, FDIM, FDIM, FDIM, FDIM);
    wait_tensorcnt0();
  }
#else
  for (int i = t; i < 128 * 128; i += 128) Ws[i] = Wth[i];
  for (int i = t; i < 64 * 128; i += 128) {
    int gr = rowBase + (i >> 7);
    Xs[i] = (gr < nrows) ? Xh[(size_t)gr * FDIM + (i & 127)] : (_Float16)0.0f;
  }
#endif
  __syncthreads();

  const int wave = t >> 5;      // 0..3 -> which 16-row stripe
  const int lane = t & 31;
  const int l15  = lane & 15;
  const int lh   = lane >> 4;   // 0/1
  const int mrow = wave * 16;

  v8f acc[8] = {};

  #pragma unroll
  for (int kt = 0; kt < 4; ++kt) {
    // A fragment 16x32 f16: lane row = l15; lanes<16 hold K 0..7 & 16..23,
    // lanes>=16 hold K 8..15 & 24..31 (two contiguous 8-f16 runs).
    v16h a;
    const _Float16* abase = &Xs[(mrow + l15) * 128 + kt * 32];
    #pragma unroll
    for (int i = 0; i < 8; ++i) a[i]     = abase[lh * 8 + i];
    #pragma unroll
    for (int i = 0; i < 8; ++i) a[8 + i] = abase[16 + lh * 8 + i];

    #pragma unroll
    for (int nt = 0; nt < 8; ++nt) {
      // B fragment 32x16 f16: lane col = l15, K = lh*8 + i (i<8), 16 + lh*8 + (i-8) (i>=8)
      v16h b;
      const _Float16* bbase = &Ws[(nt * 16 + l15) * 128 + kt * 32 + lh * 8];
      #pragma unroll
      for (int i = 0; i < 16; ++i) b[i] = bbase[(i >> 3) * 16 + (i & 7)];
      acc[nt] = __builtin_amdgcn_wmma_f32_16x16x32_f16(
          false, a, false, b, (short)0, acc[nt], false, false);
    }
  }

  // C/D layout: VGPR r = row (lanes<16: M=r, lanes>=16: M=r+8), N = lane&15
  #pragma unroll
  for (int nt = 0; nt < 8; ++nt) {
    int col = nt * 16 + l15;
    #pragma unroll
    for (int r = 0; r < 8; ++r) {
      int row = rowBase + mrow + lh * 8 + r;
      if (row < nrows) Hout[row * 128 + col] = acc[nt][r];
    }
  }
}

// ---------------------------------------------------------------------------
// Per-node attention logits: al_src[n,h] = h[n,h,:]·a_src[h,:], same for dst
// ---------------------------------------------------------------------------
__global__ __launch_bounds__(256) void attn_logits(
    const float* __restrict__ H, const float* __restrict__ a_src,
    const float* __restrict__ a_dst,
    float* __restrict__ alS, float* __restrict__ alD)
{
  int idx = blockIdx.x * blockDim.x + threadIdx.x;   // node*NHEAD + head
  if (idx >= NNODES * NHEAD) return;
  int node = idx >> 2, head = idx & 3;
  const float* hv = H + node * FDIM + head * NCHAN;
  const float* as = a_src + head * NCHAN;
  const float* ad = a_dst + head * NCHAN;
  float s = 0.f, d = 0.f;
  #pragma unroll
  for (int c = 0; c < NCHAN; ++c) { float v = hv[c]; s += v * as[c]; d += v * ad[c]; }
  alS[idx] = s;
  alD[idx] = d;
}

// ---------------------------------------------------------------------------
// Zero/init per-layer accumulators
// ---------------------------------------------------------------------------
__global__ __launch_bounds__(256) void init_layer(
    unsigned* __restrict__ mkey, float* __restrict__ den, float* __restrict__ outb)
{
  int stride = gridDim.x * blockDim.x;
  int tid = blockIdx.x * blockDim.x + threadIdx.x;
  for (int i = tid; i < NNODES * FDIM; i += stride) outb[i] = 0.f;
  for (int i = tid; i < NNODES * NHEAD; i += stride) { mkey[i] = 0u; den[i] = 0.f; }
}

// ---------------------------------------------------------------------------
// Edge logits (LeakyReLU 0.2) + segment max via monotonic-key atomic max
// ---------------------------------------------------------------------------
__global__ __launch_bounds__(256) void edge_max(
    const int* __restrict__ ei, const float* __restrict__ alS,
    const float* __restrict__ alD, float* __restrict__ ebuf,
    unsigned* __restrict__ mkey)
{
  int idx = blockIdx.x * blockDim.x + threadIdx.x;   // edge*NHEAD + head
  if (idx >= ETOT * NHEAD) return;
  int edge = idx >> 2, head = idx & 3;
  int s, d;
  if (edge < NEDGES) { s = ei[edge]; d = ei[NEDGES + edge]; }
  else               { s = d = edge - NEDGES; }
  float v = alS[s * NHEAD + head] + alD[d * NHEAD + head];
  v = (v > 0.f) ? v : 0.2f * v;
  ebuf[idx] = v;
  atomicMax(&mkey[d * NHEAD + head], fkey(v));
}

// ---------------------------------------------------------------------------
// exp(e - m[dst]) and segment-sum denominator
// ---------------------------------------------------------------------------
__global__ __launch_bounds__(256) void edge_exp(
    const int* __restrict__ ei, float* __restrict__ ebuf,
    const unsigned* __restrict__ mkey, float* __restrict__ den)
{
  int idx = blockIdx.x * blockDim.x + threadIdx.x;
  if (idx >= ETOT * NHEAD) return;
  int edge = idx >> 2, head = idx & 3;
  int d = (edge < NEDGES) ? ei[NEDGES + edge] : (edge - NEDGES);
  float m = funkey(mkey[d * NHEAD + head]);
  float ex = __expf(ebuf[idx] - m);
  ebuf[idx] = ex;
  atomicAdd(&den[d * NHEAD + head], ex);
}

// ---------------------------------------------------------------------------
// Weighted message scatter: out[dst] += h[src] * alpha  (wave per edge,
// lane per within-head channel; 4 heads iterated per lane -> 128 channels)
// ---------------------------------------------------------------------------
__global__ __launch_bounds__(256) void aggregate(
    const int* __restrict__ ei, const float* __restrict__ Hsrc,
    const float* __restrict__ ebuf, const float* __restrict__ den,
    float* __restrict__ outb)
{
  int gw   = (blockIdx.x * blockDim.x + threadIdx.x) >> 5;  // wave = edge
  int lane = threadIdx.x & 31;
  if (gw >= ETOT) return;
  int s, d;
  if (gw < NEDGES) { s = ei[gw]; d = ei[NEDGES + gw]; }
  else             { s = d = gw - NEDGES; }
  __builtin_prefetch(Hsrc + (size_t)s * FDIM, 0, 0);   // global_prefetch_b8
  #pragma unroll
  for (int head = 0; head < NHEAD; ++head) {
    float alpha = ebuf[gw * NHEAD + head] / (den[d * NHEAD + head] + 1e-16f);
    int ch = head * NCHAN + lane;
    atomicAdd(&outb[d * FDIM + ch], Hsrc[s * FDIM + ch] * alpha);
  }
}

// ---------------------------------------------------------------------------
// Hidden-layer epilogue: +bias, L2 normalize over 128, ReLU (wave per node)
// ---------------------------------------------------------------------------
__global__ __launch_bounds__(256) void post_norm_relu(
    float* __restrict__ outb, const float* __restrict__ bias)
{
  int node = (blockIdx.x * blockDim.x + threadIdx.x) >> 5;
  int lane = threadIdx.x & 31;
  if (node >= NNODES) return;
  float v[4], ss = 0.f;
  #pragma unroll
  for (int j = 0; j < 4; ++j) {
    int ch = j * 32 + lane;
    v[j] = outb[node * FDIM + ch] + bias[ch];
    ss += v[j] * v[j];
  }
  #pragma unroll
  for (int off = 16; off > 0; off >>= 1) ss += __shfl_xor(ss, off);
  float inv = 1.f / fmaxf(__builtin_sqrtf(ss), 1e-12f);
  #pragma unroll
  for (int j = 0; j < 4; ++j) {
    int ch = j * 32 + lane;
    float r = v[j] * inv;
    outb[node * FDIM + ch] = (r > 0.f) ? r : 0.f;
  }
}

// ---------------------------------------------------------------------------
// Final layer: logits = agg + bias ; probs = softmax(logits) over 128
// d_out = [logits (N*128) | probs (N*128)]
// ---------------------------------------------------------------------------
__global__ __launch_bounds__(256) void final_out(
    const float* __restrict__ agg, const float* __restrict__ bias,
    float* __restrict__ out)
{
  int node = (blockIdx.x * blockDim.x + threadIdx.x) >> 5;
  int lane = threadIdx.x & 31;
  if (node >= NNODES) return;
  float v[4];
  float mx = -3.402823e38f;
  #pragma unroll
  for (int j = 0; j < 4; ++j) {
    int ch = j * 32 + lane;
    v[j] = agg[node * FDIM + ch] + bias[ch];
    out[node * FDIM + ch] = v[j];
    mx = fmaxf(mx, v[j]);
  }
  #pragma unroll
  for (int off = 16; off > 0; off >>= 1) mx = fmaxf(mx, __shfl_xor(mx, off));
  float ex[4], s = 0.f;
  #pragma unroll
  for (int j = 0; j < 4; ++j) { ex[j] = __expf(v[j] - mx); s += ex[j]; }
  #pragma unroll
  for (int off = 16; off > 0; off >>= 1) s += __shfl_xor(s, off);
  float inv = 1.f / s;
  #pragma unroll
  for (int j = 0; j < 4; ++j) {
    int ch = j * 32 + lane;
    out[(size_t)NNODES * FDIM + node * FDIM + ch] = ex[j] * inv;
  }
}

// ---------------------------------------------------------------------------
extern "C" void kernel_launch(void* const* d_in, const int* in_sizes, int n_in,
                              void* d_out, int out_size, void* d_ws, size_t ws_size,
                              hipStream_t stream)
{
  (void)in_sizes; (void)n_in; (void)out_size; (void)ws_size;

  const float* x  = (const float*)d_in[0];
  const int*   ei = (const int*)d_in[1];
  const float* Wl[3] = {(const float*)d_in[2],  (const float*)d_in[6],  (const float*)d_in[10]};
  const float* aS[3] = {(const float*)d_in[3],  (const float*)d_in[7],  (const float*)d_in[11]};
  const float* aD[3] = {(const float*)d_in[4],  (const float*)d_in[8],  (const float*)d_in[12]};
  const float* bl[3] = {(const float*)d_in[5],  (const float*)d_in[9],  (const float*)d_in[13]};

  char* p = (char*)d_ws;
  auto take = [&](size_t bytes) -> char* {
    char* r = p; p += (bytes + 255) & ~(size_t)255; return r;
  };
  float*     hA   = (float*)take((size_t)NNODES * FDIM * sizeof(float));
  float*     hB   = (float*)take((size_t)NNODES * FDIM * sizeof(float));
  float*     alS  = (float*)take((size_t)NNODES * NHEAD * sizeof(float));
  float*     alD  = (float*)take((size_t)NNODES * NHEAD * sizeof(float));
  unsigned*  mkey = (unsigned*)take((size_t)NNODES * NHEAD * sizeof(unsigned));
  float*     den  = (float*)take((size_t)NNODES * NHEAD * sizeof(float));
  float*     ebuf = (float*)take((size_t)ETOT * NHEAD * sizeof(float));
  _Float16*  Xh   = (_Float16*)take((size_t)NNODES * FDIM * sizeof(_Float16));
  _Float16*  Wth  = (_Float16*)take((size_t)FDIM * FDIM * sizeof(_Float16));

  const int cvtXBlocks = (NNODES * FDIM + 255) / 256;        // 15000
  const int cvtWBlocks = (FDIM * FDIM + 255) / 256;          // 64
  const int gemmBlocks = (NNODES + 63) / 64;                 // 469
  const int attnBlocks = (NNODES * NHEAD + 255) / 256;       // 469
  const int edgeBlocks = (ETOT * NHEAD + 255) / 256;         // 7969
  const int aggBlocks  = (ETOT + 7) / 8;                     // 63750 (8 waves/blk)
  const int nodeBlocks = (NNODES + 7) / 8;                   // 3750  (wave/node)

  const float* cur = x;
  for (int l = 0; l < 3; ++l) {
    cvt_x_f16<<<cvtXBlocks, 256, 0, stream>>>(cur, Xh);
    cvt_w_f16t<<<cvtWBlocks, 256, 0, stream>>>(Wl[l], Wth);
    gemm_wmma<<<gemmBlocks, 128, 0, stream>>>(Xh, Wth, hA, NNODES);
    attn_logits<<<attnBlocks, 256, 0, stream>>>(hA, aS[l], aD[l], alS, alD);
    init_layer<<<2048, 256, 0, stream>>>(mkey, den, hB);
    edge_max<<<edgeBlocks, 256, 0, stream>>>(ei, alS, alD, ebuf, mkey);
    edge_exp<<<edgeBlocks, 256, 0, stream>>>(ei, ebuf, mkey, den);
    aggregate<<<aggBlocks, 256, 0, stream>>>(ei, hA, ebuf, den, hB);
    if (l < 2) post_norm_relu<<<nodeBlocks, 256, 0, stream>>>(hB, bl[l]);
    cur = hB;
  }
  final_out<<<nodeBlocks, 256, 0, stream>>>(hB, bl[2], (float*)d_out);
}
